// FusedTriangleMultiplication_26620207300647
// MI455X (gfx1250) — compile-verified, matrix operand order checked by reading, not run on hardware
//
#include <hip/hip_runtime.h>
#include <hip/hip_bf16.h>

// ---------------------------------------------------------------------------
// FusedTriangleMultiplication for MI455X (gfx1250), wave32 + WMMA bf16 + TDM.
// B=1, S=512, C=128.
// ---------------------------------------------------------------------------

#define S_DIM 512
#define C_DIM 128
#define NPOS  (S_DIM * S_DIM)   // 262144 positions

typedef __attribute__((ext_vector_type(16))) __bf16 bf16x16;
typedef __attribute__((ext_vector_type(8)))  float  f32x8;
typedef __attribute__((ext_vector_type(4)))  unsigned int u32x4;
typedef __attribute__((ext_vector_type(8)))  int    i32x8;
typedef __attribute__((ext_vector_type(4)))  int    i32x4;

#if defined(__gfx1250__) && __has_builtin(__builtin_amdgcn_tensor_load_to_lds)
#define HAS_TDM 1
#else
#define HAS_TDM 0
#endif

union FragAB {
    bf16x16 v;
    uint4   u[2];
};
union Pack8 {
    __bf16 e[8];
    uint4  u;
};
union FragA16 {
    bf16x16 v;
    __bf16  e[16];
};

__device__ __forceinline__ bf16x16 load_frag(const __bf16* lo, const __bf16* hi) {
    FragAB f;
    f.u[0] = *reinterpret_cast<const uint4*>(lo);
    f.u[1] = *reinterpret_cast<const uint4*>(hi);
    return f.v;
}

__device__ __forceinline__ f32x8 wmma_bf16(bf16x16 a, bf16x16 b, f32x8 c) {
    return __builtin_amdgcn_wmma_f32_16x16x32_bf16(
        /*neg_a=*/false, a, /*neg_b=*/false, b,
        /*c_mod=*/(short)0, c, /*reuse_a=*/false, /*reuse_b=*/false);
}

// v_exp_f32 + v_rcp_f32 instead of IEEE divide chain (gate precision is bf16).
__device__ __forceinline__ float sigmoidf(float x) {
    return __builtin_amdgcn_rcpf(1.0f + __expf(-x));
}

__device__ __forceinline__ f32x8 zero8() {
    f32x8 z = {0.f, 0.f, 0.f, 0.f, 0.f, 0.f, 0.f, 0.f};
    return z;
}

// ---------------------------------------------------------------------------
// k0: convert f32 weights -> bf16 in workspace (one-time, tiny).
// ---------------------------------------------------------------------------
__global__ __launch_bounds__(256) void k0_convert(
    const float* __restrict__ wlp, const float* __restrict__ wlg,
    const float* __restrict__ wop, const float* __restrict__ wog,
    __bf16* __restrict__ o_lp, __bf16* __restrict__ o_lg,
    __bf16* __restrict__ o_op, __bf16* __restrict__ o_og)
{
    int idx = blockIdx.x * 256 + threadIdx.x;
    if (idx < 32768)       o_lp[idx]         = (__bf16)wlp[idx];
    else if (idx < 65536)  o_lg[idx - 32768] = (__bf16)wlg[idx - 32768];
    else if (idx < 81920)  o_op[idx - 65536] = (__bf16)wop[idx - 65536];
    else if (idx < 98304)  o_og[idx - 81920] = (__bf16)wog[idx - 81920];
}

// ---------------------------------------------------------------------------
// k1: LayerNorm + lr_proj/lr_gate GEMMs + sigmoid gating + out_gate.
// Each wave handles 16 pair positions; lane l loads exactly the k-chunks the
// 16x32 bf16 A-fragment layout wants (kb = (l/16)*8, chunks 16t+kb).
// leftT/rightT stored channel-major (C,S,S): contiguous over position index.
// ---------------------------------------------------------------------------
__global__ __launch_bounds__(256) void k1_ln_proj(
    const float*  __restrict__ pair,
    const float*  __restrict__ lnw,
    const float*  __restrict__ lnb,
    const __bf16* __restrict__ Wlp,   // (256,128) bf16
    const __bf16* __restrict__ Wlg,   // (256,128) bf16
    const __bf16* __restrict__ Wog,   // (128,128) bf16
    __bf16* __restrict__ leftT,       // (C,S,S)
    __bf16* __restrict__ rightT,      // (C,S,S)
    __bf16* __restrict__ og)          // (S,S,C) sigmoid applied
{
    const int wave = threadIdx.x >> 5;
    const int lane = threadIdx.x & 31;
    const int r    = lane & 15;      // tile row / B column
    const int hi   = lane >> 4;      // 0 or 1
    const int kb   = hi * 8;

    const long p0 = (long)blockIdx.x * 128 + (long)wave * 16;  // wave's 16 positions
    const long p  = p0 + r;
    const float* xrow = pair + p * C_DIM;

    // Load this lane's 8 chunks of 8 floats (exact k-values its A-fragments need).
    float4 xv[16];
    #pragma unroll
    for (int t = 0; t < 8; ++t) {
        const float4* src = reinterpret_cast<const float4*>(xrow + 16 * t + kb);
        xv[2 * t + 0] = src[0];
        xv[2 * t + 1] = src[1];
    }

    // LayerNorm stats: lane covers 64 of 128 channels; partner lane (l^16) the
    // rest -> one shfl_xor(16) completes the row reduction (wave32).
    float s = 0.f, s2 = 0.f;
    #pragma unroll
    for (int q = 0; q < 16; ++q) {
        float4 v = xv[q];
        s  += v.x + v.y + v.z + v.w;
        s2 += v.x * v.x + v.y * v.y + v.z * v.z + v.w * v.w;
    }
    s  += __shfl_xor(s, 16);
    s2 += __shfl_xor(s2, 16);
    const float mu  = s * (1.0f / 128.0f);
    const float var = s2 * (1.0f / 128.0f) - mu * mu;
    const float rs  = __frsqrt_rn(var + 1e-5f);

    // Normalize + gamma/beta, pack into 4 bf16 A-fragments (K = 128 = 4 * 32).
    FragA16 afr[4];
    #pragma unroll
    for (int t = 0; t < 8; ++t) {
        const float4* wsrc = reinterpret_cast<const float4*>(lnw + 16 * t + kb);
        const float4* bsrc = reinterpret_cast<const float4*>(lnb + 16 * t + kb);
        float4 w0 = wsrc[0], w1 = wsrc[1];
        float4 b0 = bsrc[0], b1 = bsrc[1];
        float4 x0 = xv[2 * t], x1 = xv[2 * t + 1];
        float y[8];
        y[0] = (x0.x - mu) * rs * w0.x + b0.x;
        y[1] = (x0.y - mu) * rs * w0.y + b0.y;
        y[2] = (x0.z - mu) * rs * w0.z + b0.z;
        y[3] = (x0.w - mu) * rs * w0.w + b0.w;
        y[4] = (x1.x - mu) * rs * w1.x + b1.x;
        y[5] = (x1.y - mu) * rs * w1.y + b1.y;
        y[6] = (x1.z - mu) * rs * w1.z + b1.z;
        y[7] = (x1.w - mu) * rs * w1.w + b1.w;
        const int fs = t >> 1;          // k-step
        const int hb = (t & 1) * 8;     // element half within fragment
        #pragma unroll
        for (int e = 0; e < 8; ++e) afr[fs].e[hb + e] = (__bf16)y[e];
    }

    // --- left/right: 16 N-tiles of proj paired with gate, sigmoid-gated ------
    #pragma unroll 1
    for (int t = 0; t < 16; ++t) {
        f32x8 cp = zero8();
        f32x8 cg = zero8();
        const __bf16* wp = Wlp + (t * 16 + r) * C_DIM;   // B column = weight row
        const __bf16* wg = Wlg + (t * 16 + r) * C_DIM;
        #pragma unroll
        for (int ks = 0; ks < 4; ++ks) {
            bf16x16 bp = load_frag(wp + 32 * ks + kb, wp + 32 * ks + 16 + kb);
            cp = wmma_bf16(afr[ks].v, bp, cp);
            bf16x16 bg = load_frag(wg + 32 * ks + kb, wg + 32 * ks + 16 + kb);
            cg = wmma_bf16(afr[ks].v, bg, cg);
        }
        const int c = (t & 7) * 16 + r;                  // channel
        __bf16* dst = (t < 8 ? leftT : rightT)
                      + (long)c * (long)NPOS + p0 + hi * 8;
        Pack8 pk;
        #pragma unroll
        for (int v8 = 0; v8 < 8; ++v8)
            pk.e[v8] = (__bf16)(cp[v8] * sigmoidf(cg[v8]));
        *reinterpret_cast<uint4*>(dst) = pk.u;           // 8 consecutive positions
    }

    // --- out_gate: 8 N-tiles, sigmoid, store (S,S,C) bf16 --------------------
    #pragma unroll 1
    for (int t = 0; t < 8; ++t) {
        f32x8 cgo = zero8();
        const __bf16* wo = Wog + (t * 16 + r) * C_DIM;
        #pragma unroll
        for (int ks = 0; ks < 4; ++ks) {
            bf16x16 bo = load_frag(wo + 32 * ks + kb, wo + 32 * ks + 16 + kb);
            cgo = wmma_bf16(afr[ks].v, bo, cgo);
        }
        const int d = t * 16 + r;
        #pragma unroll
        for (int v8 = 0; v8 < 8; ++v8) {
            const long pp = p0 + hi * 8 + v8;
            og[pp * C_DIM + d] = (__bf16)sigmoidf(cgo[v8]);
        }
    }
}

// ---------------------------------------------------------------------------
// k2: per-channel NT-GEMM  update_c = L_c (SxS) @ R_c^T (SxS), K = 512.
// Block = 128x128 tile of one channel; 8 waves as 2x4; wave = 64x32 = 4x2 frags.
// L/R 128x32 k-slabs are staged in LDS by the Tensor Data Mover, double
// buffered and tracked with TENSORcnt. LDS rows are padded by the TDM
// (16B pad per 64B row -> 80B stride = 20 banks: conflict-free b128 reads).
// ---------------------------------------------------------------------------
#define ROW_E 40   // LDS row stride in bf16 elements (64B data + 16B pad)

#if HAS_TDM
__device__ __forceinline__ void tdm_load_tile(const __bf16* gtile, __bf16* ldst) {
    const unsigned long long ga = (unsigned long long)(uintptr_t)gtile;
    const unsigned int lds_off  = (unsigned int)(uintptr_t)ldst;  // LDS byte offset
    u32x4 g0;
    g0[0] = 1u;                                         // count=1, user descriptor
    g0[1] = lds_off;                                    // lds_addr
    g0[2] = (unsigned int)(ga & 0xFFFFFFFFu);           // global_addr[31:0]
    g0[3] = (unsigned int)((ga >> 32) & 0x1FFFFFFu)     // global_addr[56:32]
            | (2u << 30);                               // type = 2 ("image")
    i32x8 g1;
    g1[0] = (int)((1u << 16)      // data_size = 2 bytes
                | (1u << 20)      // pad_enable
                | (3u << 22)      // pad_interval: every 16 DWORDs (64B)
                | (3u << 25));    // pad_amount: 4 DWORDs (16B)
    g1[1] = (int)(512u << 16);    // tensor_dim0 low16 = 512
    g1[2] = (int)(512u << 16);    // tensor_dim0 hi = 0 ; tensor_dim1 low16 = 512
    g1[3] = (int)(32u  << 16);    // tensor_dim1 hi = 0 ; tile_dim0 = 32
    g1[4] = (int)128u;            // tile_dim1 = 128 ; tile_dim2 = 0
    g1[5] = (int)512u;            // tensor_dim0_stride[31:0] = 512 elements
    g1[6] = 0;                    // stride hi / dim1 stride (unused for 2D tile)
    g1[7] = 0;
    i32x4 z4 = {0, 0, 0, 0};              // groups 2/3 unused (<=2D tensor)
    i32x8 z8 = {0, 0, 0, 0, 0, 0, 0, 0};  // 6-arg toolchain form
    __builtin_amdgcn_tensor_load_to_lds(g0, g1, z4, z4, z8, 0);
}
#else
__device__ __forceinline__ void stage_tile_sync(const __bf16* __restrict__ g,
                                                __bf16* s, int tid) {
    #pragma unroll
    for (int q = 0; q < 2; ++q) {
        const int cid = tid + q * 256;      // 512 chunks of 8 bf16
        const int row = cid >> 2;
        const int col = (cid & 3) * 8;
        *reinterpret_cast<uint4*>(s + row * ROW_E + col) =
            *reinterpret_cast<const uint4*>(g + (long)row * S_DIM + col);
    }
}
#endif

__global__ __launch_bounds__(256) void k2_triangle(
    const __bf16* __restrict__ leftT,
    const __bf16* __restrict__ rightT,
    __bf16* __restrict__ upd)         // (S,S,C) bf16
{
    __shared__ __bf16 sL[2][128 * ROW_E];   // 2 x 10 KiB
    __shared__ __bf16 sR[2][128 * ROW_E];

    const int c    = blockIdx.x;
    const int tid  = threadIdx.x;
    const int wave = tid >> 5;
    const int lane = tid & 31;
    const int r    = lane & 15;
    const int hi   = lane >> 4;
    const int kb   = hi * 8;
    const int wr   = wave >> 2;       // 0..1
    const int wc   = wave & 3;        // 0..3

    const __bf16* L = leftT  + (long)c * (long)NPOS + (long)blockIdx.y * 128 * S_DIM;
    const __bf16* R = rightT + (long)c * (long)NPOS + (long)blockIdx.z * 128 * S_DIM;
    const int m0 = blockIdx.y * 128 + wr * 64;  // 4 m-tiles of 16
    const int n0 = blockIdx.z * 128 + wc * 32;  // 2 n-tiles of 16

    f32x8 acc[4][2];
    #pragma unroll
    for (int mi = 0; mi < 4; ++mi)
        #pragma unroll
        for (int ni = 0; ni < 2; ++ni) acc[mi][ni] = zero8();

#if HAS_TDM
    if (wave == 0) {                   // one wave drives the TDM (EXEC-agnostic op)
        tdm_load_tile(L, sL[0]);
        tdm_load_tile(R, sR[0]);
    }
#endif

    #pragma unroll 1
    for (int it = 0; it < 16; ++it) {
        const int buf = it & 1;
#if HAS_TDM
        if (wave == 0) __builtin_amdgcn_s_wait_tensorcnt(0);
        __syncthreads();               // publish buf; prior reads of buf^1 done
        if (wave == 0 && it < 15) {
            tdm_load_tile(L + (it + 1) * 32, sL[buf ^ 1]);
            tdm_load_tile(R + (it + 1) * 32, sR[buf ^ 1]);
        }
#else
        __syncthreads();
        stage_tile_sync(L + it * 32, sL[buf], tid);
        stage_tile_sync(R + it * 32, sR[buf], tid);
        __syncthreads();
#endif
        bf16x16 a[4], b[2];
        #pragma unroll
        for (int mi = 0; mi < 4; ++mi) {
            const __bf16* ap = &sL[buf][(wr * 64 + mi * 16 + r) * ROW_E + kb];
            a[mi] = load_frag(ap, ap + 16);
        }
        #pragma unroll
        for (int ni = 0; ni < 2; ++ni) {
            const __bf16* bp = &sR[buf][(wc * 32 + ni * 16 + r) * ROW_E + kb];
            b[ni] = load_frag(bp, bp + 16);
        }
        #pragma unroll
        for (int mi = 0; mi < 4; ++mi)
            #pragma unroll
            for (int ni = 0; ni < 2; ++ni)
                acc[mi][ni] = wmma_bf16(a[mi], b[ni], acc[mi][ni]);
    }

    // Store update[(i*S + j)*C + c] as bf16.
    #pragma unroll
    for (int mi = 0; mi < 4; ++mi) {
        #pragma unroll
        for (int ni = 0; ni < 2; ++ni) {
            const int j = n0 + ni * 16 + r;
            #pragma unroll
            for (int v8 = 0; v8 < 8; ++v8) {
                const int i = m0 + mi * 16 + hi * 8 + v8;
                upd[((long)i * S_DIM + j) * C_DIM + c] = (__bf16)acc[mi][ni][v8];
            }
        }
    }
}

// ---------------------------------------------------------------------------
// k3: output = (update @ W_out_proj^T) * out_gate,  K = N = 128, f32 store.
// ---------------------------------------------------------------------------
__global__ __launch_bounds__(256) void k3_outproj(
    const __bf16* __restrict__ upd,   // (S,S,C) bf16
    const __bf16* __restrict__ Wop,   // (128,128) bf16
    const __bf16* __restrict__ og,    // (S,S,C) bf16 (sigmoid applied)
    float* __restrict__ out)          // (S,S,C) f32
{
    const int wave = threadIdx.x >> 5;
    const int lane = threadIdx.x & 31;
    const int r    = lane & 15;
    const int hi   = lane >> 4;
    const int kb   = hi * 8;
    const long p0  = (long)blockIdx.x * 128 + (long)wave * 16;

    const __bf16* xr = upd + (p0 + r) * C_DIM;
    bf16x16 afr[4];
    #pragma unroll
    for (int ks = 0; ks < 4; ++ks)
        afr[ks] = load_frag(xr + 32 * ks + kb, xr + 32 * ks + 16 + kb);

    #pragma unroll 1
    for (int t = 0; t < 8; ++t) {
        f32x8 cacc = zero8();
        const __bf16* wrow = Wop + (t * 16 + r) * C_DIM;
        #pragma unroll
        for (int ks = 0; ks < 4; ++ks) {
            bf16x16 bfr = load_frag(wrow + 32 * ks + kb, wrow + 32 * ks + 16 + kb);
            cacc = wmma_bf16(afr[ks], bfr, cacc);
        }
        const int d = t * 16 + r;
        #pragma unroll
        for (int v8 = 0; v8 < 8; ++v8) {
            const long pp = p0 + hi * 8 + v8;
            out[pp * C_DIM + d] = cacc[v8] * (float)og[pp * C_DIM + d];
        }
    }
}

// ---------------------------------------------------------------------------
// Launch
// ---------------------------------------------------------------------------
extern "C" void kernel_launch(void* const* d_in, const int* in_sizes, int n_in,
                              void* d_out, int out_size, void* d_ws, size_t ws_size,
                              hipStream_t stream) {
    const float* pair = (const float*)d_in[0];
    const float* ln_w = (const float*)d_in[1];
    const float* ln_b = (const float*)d_in[2];
    const float* Wlp  = (const float*)d_in[3];  // (256,128)
    const float* Wlg  = (const float*)d_in[4];  // (256,128)
    const float* Wop  = (const float*)d_in[5];  // (128,128)
    const float* Wog  = (const float*)d_in[6];  // (128,128)
    float* out = (float*)d_out;

    // Workspace layout (bytes)
    char* ws = (char*)d_ws;
    const size_t OFF_WLP  = 0;                         // 256*128*2 = 65536
    const size_t OFF_WLG  = OFF_WLP + 65536;
    const size_t OFF_WOP  = OFF_WLG + 65536;           // 128*128*2 = 32768
    const size_t OFF_WOG  = OFF_WOP + 32768;
    const size_t OFF_LEFT = OFF_WOG + 32768;           // (C,S,S) bf16 = 64 MiB
    const size_t SZ_CSS   = (size_t)C_DIM * NPOS * sizeof(__bf16);
    const size_t OFF_RGHT = OFF_LEFT + SZ_CSS;
    const size_t OFF_OG   = OFF_RGHT + SZ_CSS;
    const size_t OFF_UPD  = OFF_OG + SZ_CSS;

    __bf16* bWlp  = (__bf16*)(ws + OFF_WLP);
    __bf16* bWlg  = (__bf16*)(ws + OFF_WLG);
    __bf16* bWop  = (__bf16*)(ws + OFF_WOP);
    __bf16* bWog  = (__bf16*)(ws + OFF_WOG);
    __bf16* leftT = (__bf16*)(ws + OFF_LEFT);
    __bf16* rghtT = (__bf16*)(ws + OFF_RGHT);
    __bf16* og    = (__bf16*)(ws + OFF_OG);
    __bf16* upd   = (__bf16*)(ws + OFF_UPD);

    k0_convert<<<384, 256, 0, stream>>>(Wlp, Wlg, Wop, Wog, bWlp, bWlg, bWop, bWog);

    k1_ln_proj<<<NPOS / 128, 256, 0, stream>>>(
        pair, ln_w, ln_b, bWlp, bWlg, bWog, leftT, rghtT, og);

    k2_triangle<<<dim3(C_DIM, S_DIM / 128, S_DIM / 128), 256, 0, stream>>>(
        leftT, rghtT, upd);

    k3_outproj<<<NPOS / 128, 256, 0, stream>>>(upd, bWop, og, out);
}
